// TropicalMLP_12335146074811
// MI455X (gfx1250) — matrix operand order repaired and checked by reading, not compiled, not run on hardware
//
#include <hip/hip_runtime.h>
#include <hip/hip_bf16.h>

// ---------------------------------------------------------------------------
// Tropical MLP on gfx1250.
//   trop_linear(x,W,b)[b,i] = logsumexp_j(x_bj + W_ij) + b_i
//                           = mx_b + mw_i + log( sum_j e^{x_bj-mx_b} e^{W_ij-mw_i} ) + b_i
// => elementwise exp, f16 GEMM via v_wmma_f32_16x16x32_f16, elementwise log.
// Median/IQR layernorm via per-row bitonic sort in LDS.
// ---------------------------------------------------------------------------

typedef __attribute__((ext_vector_type(16))) _Float16 v16h;
typedef __attribute__((ext_vector_type(8)))  _Float16 v8h;
typedef __attribute__((ext_vector_type(8)))  float    v8f;

#define TL_B    1024
#define TL_K    512
#define TL_H    512
#define TL_O    256

// ---------------------------------------------------------------------------
// Kernel 1: per-row max + exp.  eout[r,j] = exp(src[r,j]-max_row), mout[r]=max.
// Used for x and for each weight matrix (rows = output neurons).
// ---------------------------------------------------------------------------
__global__ void tl_rowmax_exp(const float* __restrict__ src,
                              _Float16* __restrict__ eout,
                              float* __restrict__ mout,
                              int ncols) {
    const int row = blockIdx.x;
    const int tid = threadIdx.x;
    const float* r = src + (size_t)row * ncols;

    float m = -3.4e38f;
    for (int j = tid; j < ncols; j += 256) m = fmaxf(m, r[j]);

    __shared__ float red[256];
    red[tid] = m;
    __syncthreads();
    for (int s = 128; s > 0; s >>= 1) {
        if (tid < s) red[tid] = fmaxf(red[tid], red[tid + s]);
        __syncthreads();
    }
    m = red[0];

    for (int j = tid; j < ncols; j += 256)
        eout[(size_t)row * ncols + j] = (_Float16)__expf(r[j] - m);
    if (tid == 0) mout[row] = m;
}

// ---------------------------------------------------------------------------
// Kernel 2: WMMA GEMM + log epilogue.
//   out[b,i] = mx[b] + mw[i] + log( sum_k ex[b,k]*ew[i,k] ) + bias[i]
// ex: (M x K) f16 row-major, ew: (N x K) f16 row-major (== B-matrix col-major).
// One wave per 16x16 tile, 4 waves per block; grid = (N/64, M/16).
// ---------------------------------------------------------------------------
__global__ void tl_gemm_wmma_log(const _Float16* __restrict__ ex,
                                 const float* __restrict__ mx,
                                 const _Float16* __restrict__ ew,
                                 const float* __restrict__ mw,
                                 const float* __restrict__ bias,
                                 float* __restrict__ out,
                                 int N, int K) {
    const int lane  = threadIdx.x & 31;
    const int wv    = threadIdx.x >> 5;
    const int tileN = (blockIdx.x * 4 + wv) * 16;
    const int tileM = blockIdx.y * 16;
    const int l15   = lane & 15;
    const int hi    = (lane >= 16) ? 1 : 0;

    // ISA A layout (16-bit 16x32): lanes 0-15 -> K base 0, lanes 16-31 -> K base 8,
    // two contiguous 8-half chunks (VGPR0-3 and VGPR4-7, +16 apart).
    const int abase = hi ? 8 : 0;
    // ISA B layout (16-bit 32x16): same K order as A (lanes 0-15 -> K base 0, lanes 16-31 -> K base 8, chunks +16 apart).
    const int bbase = hi ? 8 : 0;

    const _Float16* ap = ex + (size_t)(tileM + l15) * K;
    const _Float16* bp = ew + (size_t)(tileN + l15) * K;

    v8f acc = {};
#pragma unroll 4
    for (int k = 0; k < K; k += 32) {
        v8h a0 = *(const v8h*)(ap + k + abase);
        v8h a1 = *(const v8h*)(ap + k + abase + 16);
        v8h b0 = *(const v8h*)(bp + k + bbase);
        v8h b1 = *(const v8h*)(bp + k + bbase + 16);
        v16h A = __builtin_shufflevector(a0, a1, 0,1,2,3,4,5,6,7,8,9,10,11,12,13,14,15);
        v16h Bf = __builtin_shufflevector(b0, b1, 0,1,2,3,4,5,6,7,8,9,10,11,12,13,14,15);
        // (neg_a, A, neg_b, B, c_mod, C, reuse_a, reuse_b)
        acc = __builtin_amdgcn_wmma_f32_16x16x32_f16(false, A, false, Bf,
                                                     (short)0, acc, false, false);
    }

    // D layout: vgpr r, lane l : M = r + (l<16?0:8), N = l&15.
    const int n  = l15;
    const float wn = mw[tileN + n] + bias[tileN + n];
#pragma unroll
    for (int r = 0; r < 8; ++r) {
        const int gm = tileM + r + (hi ? 8 : 0);
        out[(size_t)gm * N + tileN + n] = mx[gm] + wn + __logf(acc[r]);
    }
}

// ---------------------------------------------------------------------------
// Kernel 3: tropical layernorm (median / IQR via bitonic sort in LDS)
//           + ReLU + exp-prep for the next layer.
// One block (256 thr) per row of 512.
//   med = sorted[255]; q25 = s[127]+0.75*(s[128]-s[127]); q75 = s[383]+0.25*(s[384]-s[383])
//   v = relu((h-med)/max(q75-q25,eps)*lnw + lnb);  eout = exp(v - rowmax(v)); mout = rowmax
// ---------------------------------------------------------------------------
__global__ void tl_ln_relu_prep(const float* __restrict__ h,
                                const float* __restrict__ lnw,
                                const float* __restrict__ lnb,
                                _Float16* __restrict__ eout,
                                float* __restrict__ mout) {
    const int NC  = 512;
    const int row = blockIdx.x;
    const unsigned t = threadIdx.x;

    __shared__ float s[NC];
    const float* r = h + (size_t)row * NC;
    const float x0 = r[t];
    const float x1 = r[t + 256];
    s[t] = x0;
    s[t + 256] = x1;

    // Bitonic sort, 512 elements, 256 threads (one compare pair each per round).
    for (unsigned k = 2; k <= (unsigned)NC; k <<= 1) {
        for (unsigned j = k >> 1; j > 0; j >>= 1) {
            __syncthreads();
            const unsigned i = ((t / j) * 2u * j) + (t % j);
            const bool up = ((i & k) == 0);
            const float a = s[i], b = s[i + j];
            if ((a > b) == up) { s[i] = b; s[i + j] = a; }
        }
    }
    __syncthreads();

    const float med = s[255];
    const float q25 = s[127] + 0.75f * (s[128] - s[127]);
    const float q75 = s[383] + 0.25f * (s[384] - s[383]);
    const float inv = 1.0f / fmaxf(q75 - q25, 1e-6f);

    const float v0 = fmaxf((x0 - med) * inv * lnw[t]       + lnb[t],       0.0f);
    const float v1 = fmaxf((x1 - med) * inv * lnw[t + 256] + lnb[t + 256], 0.0f);

    __syncthreads();
    s[t] = fmaxf(v0, v1);
    __syncthreads();
    for (int st = 128; st > 0; st >>= 1) {
        if ((int)t < st) s[t] = fmaxf(s[t], s[t + st]);
        __syncthreads();
    }
    const float m = s[0];

    eout[(size_t)row * NC + t]       = (_Float16)__expf(v0 - m);
    eout[(size_t)row * NC + t + 256] = (_Float16)__expf(v1 - m);
    if (t == 0) mout[row] = m;
}

// ---------------------------------------------------------------------------
// Host launcher.  Workspace layout (bytes):
// ---------------------------------------------------------------------------
extern "C" void kernel_launch(void* const* d_in, const int* in_sizes, int n_in,
                              void* d_out, int out_size, void* d_ws, size_t ws_size,
                              hipStream_t stream) {
    (void)in_sizes; (void)n_in; (void)out_size; (void)ws_size;

    const float* x    = (const float*)d_in[0];
    const float* w1   = (const float*)d_in[1];
    const float* b1   = (const float*)d_in[2];
    const float* ln1w = (const float*)d_in[3];
    const float* ln1b = (const float*)d_in[4];
    const float* w2   = (const float*)d_in[5];
    const float* b2   = (const float*)d_in[6];
    const float* ln2w = (const float*)d_in[7];
    const float* ln2b = (const float*)d_in[8];
    const float* w3   = (const float*)d_in[9];
    const float* b3   = (const float*)d_in[10];
    float* out = (float*)d_out;

    char* ws = (char*)d_ws;
    size_t off = 0;
    auto carve = [&](size_t bytes) { char* p = ws + off; off = (off + bytes + 255) & ~(size_t)255; return p; };

    _Float16* ew1 = (_Float16*)carve((size_t)TL_H * TL_K * 2);
    _Float16* ew2 = (_Float16*)carve((size_t)TL_H * TL_H * 2);
    _Float16* ew3 = (_Float16*)carve((size_t)TL_O * TL_H * 2);
    float*    mw1 = (float*)carve(TL_H * 4);
    float*    mw2 = (float*)carve(TL_H * 4);
    float*    mw3 = (float*)carve(TL_O * 4);
    float*    mxA = (float*)carve(TL_B * 4);
    float*    mxB = (float*)carve(TL_B * 4);
    _Float16* exA = (_Float16*)carve((size_t)TL_B * TL_K * 2);
    _Float16* exB = (_Float16*)carve((size_t)TL_B * TL_H * 2);
    float*    hbf = (float*)carve((size_t)TL_B * TL_H * 4);

    const dim3 blk256(256), blkG(128);

    // Prep: exp(weights - rowmax), exp(x - rowmax)
    tl_rowmax_exp<<<dim3(TL_H), blk256, 0, stream>>>(w1, ew1, mw1, TL_K);
    tl_rowmax_exp<<<dim3(TL_H), blk256, 0, stream>>>(w2, ew2, mw2, TL_H);
    tl_rowmax_exp<<<dim3(TL_O), blk256, 0, stream>>>(w3, ew3, mw3, TL_H);
    tl_rowmax_exp<<<dim3(TL_B), blk256, 0, stream>>>(x,  exA, mxA, TL_K);

    // Layer 1: h = troplin(x, w1, b1)
    tl_gemm_wmma_log<<<dim3(TL_H / 64, TL_B / 16), blkG, 0, stream>>>(
        exA, mxA, ew1, mw1, b1, hbf, TL_H, TL_K);
    // LN + ReLU + exp-prep for layer 2
    tl_ln_relu_prep<<<dim3(TL_B), blk256, 0, stream>>>(hbf, ln1w, ln1b, exB, mxB);

    // Layer 2
    tl_gemm_wmma_log<<<dim3(TL_H / 64, TL_B / 16), blkG, 0, stream>>>(
        exB, mxB, ew2, mw2, b2, hbf, TL_H, TL_H);
    tl_ln_relu_prep<<<dim3(TL_B), blk256, 0, stream>>>(hbf, ln2w, ln2b, exA, mxA);

    // Layer 3 -> d_out (1024 x 256 f32)
    tl_gemm_wmma_log<<<dim3(TL_O / 64, TL_B / 16), blkG, 0, stream>>>(
        exA, mxA, ew3, mw3, b3, out, TL_O, TL_H);
}
